// MaskedVanillaAttention_34952443855042
// MI455X (gfx1250) — matrix operand order repaired and checked by reading, not compiled, not run on hardware
//
#include <hip/hip_runtime.h>
#include <hip/hip_bf16.h>

// MI455X / gfx1250, wave32. All matrix contractions via v_wmma_f32_16x16x32_f16.

typedef __attribute__((ext_vector_type(16))) _Float16 v16h;
typedef __attribute__((ext_vector_type(8)))  float    v8f;
typedef __attribute__((ext_vector_type(4)))  float    v4f;   // native vector for NT stores

#define WMMA16(a, b, c) \
  __builtin_amdgcn_wmma_f32_16x16x32_f16(false, (a), false, (b), (short)0, (c), false, false)

#define BS 8
#define QL 512
#define SL 2048
#define DM 256
#define NH 4
#define HD 64

// ---------------- fragment loaders (CDNA5 ISA 7.12.2 layouts) ----------------

// A-matrix 16x32 (MxK), f32 source row-major with leading dim lda.
// lanes 0-15: K = {0..7, 16..23}; lanes 16-31: K = {8..15, 24..31}; row M = lane&15.
__device__ __forceinline__ v16h load_a_frag_f32(const float* __restrict__ A, int lda, int lane) {
  const int m    = lane & 15;
  const int koff = (lane & 16) ? 8 : 0;
  const float* r = A + m * lda + koff;
  float4 x0 = *(const float4*)(r);
  float4 x1 = *(const float4*)(r + 4);
  float4 x2 = *(const float4*)(r + 16);
  float4 x3 = *(const float4*)(r + 20);
  v16h a;
  a[0]=(_Float16)x0.x; a[1]=(_Float16)x0.y; a[2]=(_Float16)x0.z; a[3]=(_Float16)x0.w;
  a[4]=(_Float16)x1.x; a[5]=(_Float16)x1.y; a[6]=(_Float16)x1.z; a[7]=(_Float16)x1.w;
  a[8]=(_Float16)x2.x; a[9]=(_Float16)x2.y; a[10]=(_Float16)x2.z; a[11]=(_Float16)x2.w;
  a[12]=(_Float16)x3.x; a[13]=(_Float16)x3.y; a[14]=(_Float16)x3.z; a[15]=(_Float16)x3.w;
  return a;
}

// B-matrix 32x16 (KxN) sourced from a transposed store S[n][k] (leading dim ld):
// B[k][n] = S[n][k]. lane n = lane&15; lanes 0-15: K = {0..7, 16..23}; lanes 16-31: K = {8..15, 24..31}.
// Works for global or LDS-resident S (inlined; LDS provenance -> ds_load_b128).
__device__ __forceinline__ v16h load_bT_frag_f32(const float* S, int ld, int lane) {
  const int n  = lane & 15;
  const int kb = (lane & 16) ? 8 : 0;
  const float* r = S + n * ld + kb;
  float4 x0 = *(const float4*)(r);
  float4 x1 = *(const float4*)(r + 4);
  float4 x2 = *(const float4*)(r + 16);
  float4 x3 = *(const float4*)(r + 20);
  v16h b;
  b[0]=(_Float16)x0.x; b[1]=(_Float16)x0.y; b[2]=(_Float16)x0.z; b[3]=(_Float16)x0.w;
  b[4]=(_Float16)x1.x; b[5]=(_Float16)x1.y; b[6]=(_Float16)x1.z; b[7]=(_Float16)x1.w;
  b[8]=(_Float16)x2.x; b[9]=(_Float16)x2.y; b[10]=(_Float16)x2.z; b[11]=(_Float16)x2.w;
  b[12]=(_Float16)x3.x; b[13]=(_Float16)x3.y; b[14]=(_Float16)x3.z; b[15]=(_Float16)x3.w;
  return b;
}

// C/D 16x16 f32: VGPR r holds row (lane<16 ? r : 8+r), column = lane&15.
__device__ __forceinline__ void store_d_tile(float* __restrict__ C, int ldc, int lane, v8f c) {
  const int n  = lane & 15;
  const int mo = (lane & 16) ? 8 : 0;
#pragma unroll
  for (int r = 0; r < 8; ++r) C[(size_t)(mo + r) * ldc + n] = c[r];
}

// ---------------- generic Y = X @ W^T (torch Linear) ----------------
__global__ void __launch_bounds__(128)
gemm_xwT_kernel(const float* __restrict__ X, const float* __restrict__ W,
                float* __restrict__ Y, int M, int N, int K) {
  const int lane = threadIdx.x & 31;
  const int wave = blockIdx.x * (blockDim.x >> 5) + (threadIdx.x >> 5);
  const int ntn  = N >> 4;
  const int mt   = wave / ntn;
  const int nt   = wave - mt * ntn;
  if (mt * 16 >= M) return;
  const float* Arow = X + (size_t)(mt * 16) * K;
  const float* Srow = W + (size_t)(nt * 16) * K;
  v8f c = {0.f, 0.f, 0.f, 0.f, 0.f, 0.f, 0.f, 0.f};
#pragma unroll 4
  for (int k0 = 0; k0 < K; k0 += 32) {
    v16h a = load_a_frag_f32(Arow + k0, K, lane);
    v16h b = load_bT_frag_f32(Srow + k0, K, lane);
    c = WMMA16(a, b, c);
  }
  store_d_tile(Y + (size_t)(mt * 16) * N + nt * 16, N, lane, c);
}

// ---------------- V projection with per-head transposed output ----------------
// VT[b, h, d, s] = (v @ w_v^T)[b, s, h*64+d]. Transposed store makes each lane's
// 8 D-values contiguous in s -> two b128 stores, and makes the attention P@V
// B-fragments contiguous loads. M = 16384 rows (b*2048+s), N = 256, K = 256.
__global__ void __launch_bounds__(128)
gemm_xwT_vT_kernel(const float* __restrict__ X, const float* __restrict__ W,
                   float* __restrict__ VT) {
  const int lane = threadIdx.x & 31;
  const int wave = blockIdx.x * (blockDim.x >> 5) + (threadIdx.x >> 5);
  const int mt   = wave >> 4;
  const int nt   = wave & 15;
  const float* Arow = X + (size_t)(mt * 16) * DM;
  const float* Srow = W + (size_t)(nt * 16) * DM;
  v8f c = {0.f, 0.f, 0.f, 0.f, 0.f, 0.f, 0.f, 0.f};
#pragma unroll 4
  for (int k0 = 0; k0 < DM; k0 += 32) {
    v16h a = load_a_frag_f32(Arow + k0, DM, lane);
    v16h b = load_bT_frag_f32(Srow + k0, DM, lane);
    c = WMMA16(a, b, c);
  }
  const int n_g = nt * 16 + (lane & 15);             // column = h*64 + d
  const int m0  = mt * 16 + ((lane & 16) ? 8 : 0);   // first of 8 consecutive rows
  const int bb  = m0 >> 11;                          // 2048 key rows per batch
  const int ss  = m0 & (SL - 1);
  float* dst = VT + (((size_t)(bb * NH + (n_g >> 6))) * HD + (n_g & 63)) * SL + ss;
  float4 t0 = {c[0], c[1], c[2], c[3]};
  float4 t1 = {c[4], c[5], c[6], c[7]};
  *(float4*)dst = t0;
  *(float4*)(dst + 4) = t1;
}

// ---------------- qt[b,h,q,i] = sum_j qh[b,q,h*64+j] * E_b[i,j] ----------------
__global__ void __launch_bounds__(128)
qspa_kernel(const float* __restrict__ qh, const float* __restrict__ E,
            float* __restrict__ qt) {
  const int lane = threadIdx.x & 31;
  int wave = blockIdx.x * (blockDim.x >> 5) + (threadIdx.x >> 5);
  const int qtile = wave & 31; wave >>= 5;
  const int h = wave & 3;
  const int b = wave >> 2;
  const float* Abase = qh + ((size_t)(b * QL + qtile * 16)) * DM + h * HD;
  const float* Eb    = E + (size_t)b * (HD * HD);
  float*       outp  = qt + (((size_t)(b * NH + h)) * QL + qtile * 16) * HD;
#pragma unroll
  for (int nt = 0; nt < 4; ++nt) {
    v8f c = {0.f, 0.f, 0.f, 0.f, 0.f, 0.f, 0.f, 0.f};
#pragma unroll
    for (int k0 = 0; k0 < HD; k0 += 32) {
      v16h a  = load_a_frag_f32(Abase + k0, DM, lane);
      v16h bf = load_bT_frag_f32(Eb + (size_t)(nt * 16) * HD + k0, HD, lane);
      c = WMMA16(a, bf, c);
    }
    store_d_tile(outp + nt * 16, HD, lane, c);
  }
}

// ---------------- kt[b,h,s,i] = sum_j E_{b,s}[i,j] * kh[b,s,h*64+j] ----------------
// One wave per key position: D(64x16, 4 cols used) = E(64x64) x KH(64x16).
// The 268 MB kv_spa_embed stream dominates the whole problem's HBM traffic.
__global__ void __launch_bounds__(128)
kspa_kernel(const float* __restrict__ kh, const float* __restrict__ Es,
            float* __restrict__ kt) {
  const int lane = threadIdx.x & 31;
  int wave = blockIdx.x * (blockDim.x >> 5) + (threadIdx.x >> 5);
  const int s = wave & (SL - 1);
  const int b = wave >> 11;
  const float* E     = Es + ((size_t)(b * SL + s)) * (HD * HD);
  const float* khrow = kh + ((size_t)(b * SL + s)) * DM;   // S[n=h][k=j] = khrow[n*64+j]
  __builtin_prefetch(E + HD * HD, 0, 1);                   // next position's 16KB matrix
  const int n  = lane & 15;
  const int mo = (lane & 16) ? 8 : 0;
  v16h bfr0 = load_bT_frag_f32(khrow, HD, lane);           // j = 0..31
  v16h bfr1 = load_bT_frag_f32(khrow + 32, HD, lane);      // j = 32..63
#pragma unroll
  for (int mt = 0; mt < 4; ++mt) {
    v16h a0 = load_a_frag_f32(E + (size_t)(mt * 16) * HD, HD, lane);
    v16h a1 = load_a_frag_f32(E + (size_t)(mt * 16) * HD + 32, HD, lane);
    v8f c = {0.f, 0.f, 0.f, 0.f, 0.f, 0.f, 0.f, 0.f};
    c = WMMA16(a0, bfr0, c);
    c = WMMA16(a1, bfr1, c);
    if (n < NH) {                                          // only 4 head columns are real
      float* dst = kt + (((size_t)(b * NH + n)) * SL + s) * HD + mt * 16 + mo;
#pragma unroll
      for (int r = 0; r < 8; ++r) dst[r] = c[r];
    }
  }
}

// ---------------- fused scores + softmax(mask,bias) + attn output + O = P@V ----------------
// One wave per (b,h,16-row q tile); 4 waves/block share the same (b,h), so K-tiles
// (pass 1) and V-tiles (pass 2) are staged once per block into LDS (320KB/WGP) and
// consumed by all 4 waves: 4x cut in K/V L2 traffic, fragment reads become ds_load_b128.
// LDS rows padded (68/36 floats) so the 16 fragment lanes hit distinct banks.
// Raw biased scores live in the attn output region between passes (L2-hot, 192MB L2);
// final normalized attn is written nontemporal (never re-read).
__global__ void __launch_bounds__(128)
attn_kernel(const float* __restrict__ qt, const float* __restrict__ kt,
            const float* __restrict__ vt, const int* __restrict__ mask,
            const float* __restrict__ bias, float* __restrict__ attn,
            float* __restrict__ o_tmp) {
  __shared__ float s_m[4][16];
  __shared__ float s_l[4][16];
  __shared__ float sK[16 * 68];   // 16 key rows x 64 i-cols (+4 pad)
  __shared__ float sV[64 * 36];   // 64 d rows x 32 s-cols (+4 pad)
  const int tid  = threadIdx.x;
  const int lane = tid & 31;
  const int wid  = tid >> 5;
  int wave = blockIdx.x * 4 + wid;
  const int qtile = wave & 31; wave >>= 5;   // 32 qtiles: blocks never straddle (b,h)
  const int h = wave & 3;
  const int b = wave >> 2;

  const float scale = 0.125f;  // 1/sqrt(64)
  const float* Aq   = qt + (((size_t)(b * NH + h)) * QL + qtile * 16) * HD;
  const float* Kt   = kt + ((size_t)(b * NH + h)) * SL * HD;
  const float* Vt   = vt + ((size_t)(b * NH + h)) * HD * SL;
  const int*   mrow = mask + (size_t)b * SL;
  const float* brow = bias + (size_t)b * SL;
  float*       Prow = attn + (((size_t)(b * NH + h)) * QL + qtile * 16) * SL;

  const int n  = lane & 15;
  const int mo = (lane & 16) ? 8 : 0;

  float rmax[8], rsum[8];
#pragma unroll
  for (int r = 0; r < 8; ++r) { rmax[r] = -3.0e38f; rsum[r] = 0.f; }

  v16h aq0 = load_a_frag_f32(Aq, HD, lane);        // reused across all 128 s-tiles
  v16h aq1 = load_a_frag_f32(Aq + 32, HD, lane);

  // ---- pass 1: scores + online softmax stats ----
  for (int s0 = 0; s0 < SL; s0 += 16) {
    __syncthreads();                               // previous tile's readers done
#pragma unroll
    for (int j = 0; j < 2; ++j) {                  // 1024 floats = 256 float4, coop
      const int i = tid + j * 128;
      const int row = i >> 4, cc = (i & 15) << 2;
      *(float4*)(sK + row * 68 + cc) = *(const float4*)(Kt + (size_t)(s0 + row) * HD + cc);
    }
    __syncthreads();
    v16h b0 = load_bT_frag_f32(sK, 68, lane);      // i = 0..31
    v16h b1 = load_bT_frag_f32(sK + 32, 68, lane); // i = 32..63
    v8f c = {0.f, 0.f, 0.f, 0.f, 0.f, 0.f, 0.f, 0.f};
    c = WMMA16(aq0, b0, c);
    c = WMMA16(aq1, b1, c);
    const int s = s0 + n;
    const float addv = (float)mrow[s] * -1.0e9f - brow[s];
#pragma unroll
    for (int r = 0; r < 8; ++r) c[r] = c[r] * scale + addv;
#pragma unroll
    for (int r = 0; r < 8; ++r) Prow[(size_t)(mo + r) * SL + s] = c[r];  // raw, coalesced
#pragma unroll
    for (int r = 0; r < 8; ++r) {                  // online stats per row (16-lane halves)
      float tmax = c[r];
#pragma unroll
      for (int off = 1; off < 16; off <<= 1) tmax = fmaxf(tmax, __shfl_xor(tmax, off, 32));
      const float newm = fmaxf(rmax[r], tmax);
      float pe = __expf(c[r] - newm);
#pragma unroll
      for (int off = 1; off < 16; off <<= 1) pe += __shfl_xor(pe, off, 32);
      rsum[r] = rsum[r] * __expf(rmax[r] - newm) + pe;
      rmax[r] = newm;
    }
  }

  // D-layout stats -> per-row table in LDS for A-layout pass 2
  if (n == 0) {
#pragma unroll
    for (int r = 0; r < 8; ++r) { s_m[wid][mo + r] = rmax[r]; s_l[wid][mo + r] = rsum[r]; }
  }
  __syncthreads();
  const float rowm = s_m[wid][lane & 15];
  const float invl = 1.0f / s_l[wid][lane & 15];

  // ---- pass 2: renormalize attn + O = P @ V ----
  const v8f vzero = {0.f, 0.f, 0.f, 0.f, 0.f, 0.f, 0.f, 0.f};
  v8f o_acc[4];
#pragma unroll
  for (int d = 0; d < 4; ++d) o_acc[d] = vzero;

  const int mA    = lane & 15;
  const int koffA = (lane & 16) ? 8 : 0;
  float* PA = Prow + (size_t)mA * SL;
  for (int s0 = 0; s0 < SL; s0 += 32) {
    __syncthreads();
#pragma unroll
    for (int j = 0; j < 4; ++j) {                  // 2048 floats = 512 float4, coop
      const int i = tid + j * 128;
      const int row = i >> 3, cc = (i & 7) << 2;
      *(float4*)(sV + row * 36 + cc) = *(const float4*)(Vt + (size_t)row * SL + s0 + cc);
    }
    __syncthreads();

    float xv[16];
    float* p0 = PA + s0 + koffA;
    float* p1 = PA + s0 + 16 + koffA;
    { float4 t0 = *(const float4*)p0;       float4 t1 = *(const float4*)(p0 + 4);
      xv[0]=t0.x; xv[1]=t0.y; xv[2]=t0.z; xv[3]=t0.w;
      xv[4]=t1.x; xv[5]=t1.y; xv[6]=t1.z; xv[7]=t1.w; }
    { float4 t0 = *(const float4*)p1;       float4 t1 = *(const float4*)(p1 + 4);
      xv[8]=t0.x;  xv[9]=t0.y;  xv[10]=t0.z; xv[11]=t0.w;
      xv[12]=t1.x; xv[13]=t1.y; xv[14]=t1.z; xv[15]=t1.w; }
    v16h pa;
#pragma unroll
    for (int t = 0; t < 16; ++t) {
      const float e = __expf(xv[t] - rowm) * invl;
      xv[t] = e;
      pa[t] = (_Float16)e;
    }
    { v4f t0 = {xv[0], xv[1], xv[2], xv[3]};   v4f t1 = {xv[4], xv[5], xv[6], xv[7]};
      __builtin_nontemporal_store(t0, (v4f*)p0);
      __builtin_nontemporal_store(t1, (v4f*)(p0 + 4)); }
    { v4f t0 = {xv[8], xv[9], xv[10], xv[11]}; v4f t1 = {xv[12], xv[13], xv[14], xv[15]};
      __builtin_nontemporal_store(t0, (v4f*)p1);
      __builtin_nontemporal_store(t1, (v4f*)(p1 + 4)); }
#pragma unroll
    for (int d = 0; d < 4; ++d) {
      v16h bv = load_bT_frag_f32(sV + (d * 16) * 36, 36, lane);
      o_acc[d] = WMMA16(pa, bv, o_acc[d]);
    }
  }

  // torch-faithful scramble: o.transpose(2,3).reshape -> o_tmp[b][(h*64+d)*512 + q]
  float* ob = o_tmp + (size_t)b * (QL * DM);
#pragma unroll
  for (int d = 0; d < 4; ++d) {
    const int dc = d * 16 + n;
#pragma unroll
    for (int r = 0; r < 8; ++r)
      ob[(size_t)(h * HD + dc) * QL + (qtile * 16 + mo + r)] = o_acc[d][r];
  }
}

// ---------------- launcher ----------------
extern "C" void kernel_launch(void* const* d_in, const int* in_sizes, int n_in,
                              void* d_out, int out_size, void* d_ws, size_t ws_size,
                              hipStream_t stream) {
  (void)in_sizes; (void)n_in; (void)out_size; (void)ws_size;
  const float* q    = (const float*)d_in[0];
  const float* k    = (const float*)d_in[1];
  const float* v    = (const float*)d_in[2];
  const float* qse  = (const float*)d_in[3];
  const float* kse  = (const float*)d_in[4];
  const int*   mask = (const int*)d_in[5];
  const float* bias = (const float*)d_in[6];
  const float* w_q  = (const float*)d_in[7];
  const float* w_k  = (const float*)d_in[8];
  const float* w_v  = (const float*)d_in[9];
  const float* w_o  = (const float*)d_in[10];

  float* out  = (float*)d_out;
  float* attn = out + (size_t)BS * QL * DM;          // second tuple output

  float* ws = (float*)d_ws;                          // ~63 MB of f32 scratch
  float* qh = ws;                                    // [8*512, 256]
  float* kh = qh + (size_t)BS * QL * DM;             // [8*2048, 256]
  float* vt = kh + (size_t)BS * SL * DM;             // [8,4,64,2048] (transposed V)
  float* qt = vt + (size_t)BS * NH * HD * SL;        // [8,4,512,64]
  float* kt = qt + (size_t)BS * NH * QL * HD;        // [8,4,2048,64]
  float* ot = kt + (size_t)BS * NH * SL * HD;        // [8, 512*256] scrambled O

  const dim3 blk(128);                               // 4 waves / block
  gemm_xwT_kernel<<<1024, blk, 0, stream>>>(q, w_q, qh, BS * QL, DM, DM);
  gemm_xwT_kernel<<<4096, blk, 0, stream>>>(k, w_k, kh, BS * SL, DM, DM);
  gemm_xwT_vT_kernel<<<4096, blk, 0, stream>>>(v, w_v, vt);
  qspa_kernel<<<256, blk, 0, stream>>>(qh, qse, qt);
  kspa_kernel<<<4096, blk, 0, stream>>>(kh, kse, kt);
  attn_kernel<<<256, blk, 0, stream>>>(qt, kt, vt, mask, bias, attn, ot);
  gemm_xwT_kernel<<<1024, blk, 0, stream>>>(ot, w_o, out, BS * QL, DM, DM);
}
